// ConditionalFeedForward_9534827397919
// MI455X (gfx1250) — matrix operand order repaired and checked by reading, not compiled, not run on hardware
//
#include <hip/hip_runtime.h>
#include <hip/hip_bf16.h>

// ---- problem constants (match reference) ----
#define DIM     1024
#define INTER   2048
#define NEXP    8
#define MTOK    1024
#define TOPK    2
#define TSLOTS  (MTOK * TOPK)   // 2048 row-slots

// LDS panel stride: 1024 bf16 + 8 pad -> row stride = 516 dwords = 4-bank skew,
// so a 16-row column-group b128 read is bank-conflict free.
#define LDSTRIDE 1032

typedef __attribute__((ext_vector_type(16))) __bf16 v16bf;
typedef __attribute__((ext_vector_type(8)))  float  v8f;
typedef int v4i __attribute__((__vector_size__(4 * sizeof(int))));  // matches builtin proto

struct BF16x16 { __bf16 v[16]; };   // 32 bytes == v16bf
struct BF16x4  { __bf16 v[4];  };   // 8 bytes

#if __has_builtin(__builtin_amdgcn_global_load_async_to_lds_b128)
#define HAVE_ASYNC_LDS 1
#endif

__device__ __forceinline__ void wait_asynccnt0() {
#if __has_builtin(__builtin_amdgcn_s_wait_asynccnt)
    __builtin_amdgcn_s_wait_asynccnt(0);
#else
    asm volatile("s_wait_asynccnt 0x0" ::: "memory");
#endif
}

// ---- A fragment: 16x32 bf16 from an LDS panel -------------------------------
// Layout (matches B): lanes 0-15 hold M=lane, K=0..15; lanes 16-31 hold
// M=lane-16, K=16..31.
__device__ __forceinline__ v16bf load_a_frag(const __bf16* lds, int lane, int k0) {
    const int m     = lane & 15;
    const int kbase = (lane >> 4) << 4;           // 0 or 16
    const __bf16* p = lds + (size_t)m * LDSTRIDE + k0 + kbase;
    BF16x16 f;
    *(uint4*)&f.v[0] = *(const uint4*)(p);        // K = kbase .. kbase+7
    *(uint4*)&f.v[8] = *(const uint4*)(p + 8);    // K = kbase+8 .. kbase+15
    return __builtin_bit_cast(v16bf, f);
}

// ---- B fragment: 32x16 from one fp32 weight row -----------------------------
// B[k][n] = W[row n][k]; lane L covers column n0+(L&15); lanes 0-15 hold
// K=0..15, lanes 16-31 hold K=16..31 (caller bakes khalf*16 into wrow).
__device__ __forceinline__ v16bf load_b_frag_f32(const float* __restrict__ wrow) {
    BF16x16 f;
#pragma unroll
    for (int i = 0; i < 16; i += 4) {
        float4 t = *(const float4*)(wrow + i);
        f.v[i + 0] = (__bf16)t.x;
        f.v[i + 1] = (__bf16)t.y;
        f.v[i + 2] = (__bf16)t.z;
        f.v[i + 3] = (__bf16)t.w;
    }
    return __builtin_bit_cast(v16bf, f);
}

// fast SiLU: g * sigmoid(g) with v_rcp_f32 (avoids IEEE div expansion)
__device__ __forceinline__ float silu_fast(float g) {
    return g * __builtin_amdgcn_rcpf(1.f + __expf(-g));
}

// ---- routing ---------------------------------------------------------------
__global__ void moe_zero_counts(int* __restrict__ counts) {
    if (threadIdx.x < NEXP) counts[threadIdx.x] = 0;
}

__global__ void moe_route(const int* __restrict__ expert_idx,
                          int* __restrict__ counts,
                          int* __restrict__ bucket) {
    int t = blockIdx.x * blockDim.x + threadIdx.x;
    if (t < TSLOTS) {
        int e    = expert_idx[t];
        int slot = atomicAdd(&counts[e], 1);
        bucket[e * TSLOTS + slot] = t;   // order varies; result is order-invariant
    }
}

// ---- GEMM1: h = silu(x @ w1^T) * (x @ w3^T), bf16 out ----------------------
// grid.x = NEXP*128 (expert, m-tile of 16 slots), grid.y = INTER/128.
// 256 threads = 8 waves; each wave: one 16x16 gate tile + one 16x16 up tile.
__global__ __launch_bounds__(256) void moe_gemm1(
        const float* __restrict__ x,
        const int*   __restrict__ bucket,
        const int*   __restrict__ counts,
        const float* __restrict__ w13,
        __bf16*      __restrict__ hbuf) {
    __shared__ __bf16 sx[16 * LDSTRIDE];
    __shared__ int    s_t[16];

    const int e     = blockIdx.x >> 7;
    const int tile0 = (blockIdx.x & 127) << 4;
    const int count = counts[e];
    if (tile0 >= count) return;                    // block-uniform

    if (threadIdx.x < 16) {
        int slot = tile0 + (int)threadIdx.x;
        s_t[threadIdx.x] = bucket[e * TSLOTS + min(slot, count - 1)];
    }
    __syncthreads();

    // stage gathered 16xD x-panel as bf16, float4 -> packed bf16x4 (8B) chunks
    for (int i = threadIdx.x; i < 16 * (DIM / 4); i += 256) {
        int r = i >> 8;                            // DIM/4 = 256 chunks per row
        int c = (i & 255) * 4;
        int tok = s_t[r] / TOPK;                   // flat slot -> token row
        float4 t = *(const float4*)&x[(size_t)tok * DIM + c];
        BF16x4 b;
        b.v[0] = (__bf16)t.x; b.v[1] = (__bf16)t.y;
        b.v[2] = (__bf16)t.z; b.v[3] = (__bf16)t.w;
        *(BF16x4*)&sx[r * LDSTRIDE + c] = b;
    }
    __syncthreads();

    const int lane  = threadIdx.x & 31;
    const int wave  = threadIdx.x >> 5;
    const int ncol  = lane & 15;
    const int khalf = lane >> 4;
    const int n     = blockIdx.y * 128 + wave * 16 + ncol;   // h column in [0, INTER)
    const float* wg = w13 + ((size_t)e * 2 * INTER + n) * DIM + khalf * 16;          // gate row n
    const float* wu = w13 + ((size_t)e * 2 * INTER + INTER + n) * DIM + khalf * 16;  // up row n+I

    // cache this lane's output row ids in registers (avoid ds loads in epilogue)
    int trow[8];
#pragma unroll
    for (int i = 0; i < 8; ++i) trow[i] = s_t[i + khalf * 8];

    v8f acc_g = {0.f, 0.f, 0.f, 0.f, 0.f, 0.f, 0.f, 0.f};
    v8f acc_u = {0.f, 0.f, 0.f, 0.f, 0.f, 0.f, 0.f, 0.f};

    for (int k0 = 0; k0 < DIM; k0 += 32) {
        __builtin_prefetch(wg + k0 + 32, 0, 1);
        __builtin_prefetch(wu + k0 + 32, 0, 1);
        v16bf a  = load_a_frag(sx, lane, k0);
        v16bf bg = load_b_frag_f32(wg + k0);
        v16bf bu = load_b_frag_f32(wu + k0);
        acc_g = __builtin_amdgcn_wmma_f32_16x16x32_bf16(false, a, false, bg,
                                                        (short)0, acc_g, false, false);
        acc_u = __builtin_amdgcn_wmma_f32_16x16x32_bf16(false, a, false, bu,
                                                        (short)0, acc_u, false, false);
    }

    // fused SiLU(gate)*up, scatter bf16 (C layout: M = i + 8*khalf, N = ncol)
    __bf16 hv[8];
#pragma unroll
    for (int i = 0; i < 8; ++i) hv[i] = (__bf16)(silu_fast(acc_g[i]) * acc_u[i]);

    if (tile0 + 16 <= count) {                     // full tile: no masking
#pragma unroll
        for (int i = 0; i < 8; ++i)
            hbuf[(size_t)trow[i] * INTER + n] = hv[i];
    } else {
#pragma unroll
        for (int i = 0; i < 8; ++i)
            if (tile0 + i + khalf * 8 < count)
                hbuf[(size_t)trow[i] * INTER + n] = hv[i];
    }
}

// ---- GEMM2: out = h @ w2^T --------------------------------------------------
// grid.x = NEXP*128, grid.y = DIM/128. K = INTER staged as two 16x1024 panels,
// copied with gfx1250 async LDS loads when available.
__global__ __launch_bounds__(256) void moe_gemm2(
        const __bf16* __restrict__ hbuf,
        const int*    __restrict__ bucket,
        const int*    __restrict__ counts,
        const float*  __restrict__ w2,
        float*        __restrict__ out) {
    __shared__ __bf16 sh[16 * LDSTRIDE];
    __shared__ int    s_t[16];

    const int e     = blockIdx.x >> 7;
    const int tile0 = (blockIdx.x & 127) << 4;
    const int count = counts[e];
    if (tile0 >= count) return;

    if (threadIdx.x < 16) {
        int slot = tile0 + (int)threadIdx.x;
        s_t[threadIdx.x] = bucket[e * TSLOTS + min(slot, count - 1)];
    }
    __syncthreads();

    const int lane  = threadIdx.x & 31;
    const int wave  = threadIdx.x >> 5;
    const int ncol  = lane & 15;
    const int khalf = lane >> 4;
    const int d     = blockIdx.y * 128 + wave * 16 + ncol;   // out column
    const float* wr = w2 + ((size_t)e * DIM + d) * INTER + khalf * 16;

    int trow[8];
#pragma unroll
    for (int i = 0; i < 8; ++i) trow[i] = s_t[i + khalf * 8];

    v8f acc = {0.f, 0.f, 0.f, 0.f, 0.f, 0.f, 0.f, 0.f};

    for (int kp = 0; kp < INTER; kp += 1024) {
        if (kp) __syncthreads();                   // panel reuse fence

        // stage 16x1024 bf16 panel as 16B chunks (128 chunks/row)
#ifdef HAVE_ASYNC_LDS
        for (int i = threadIdx.x; i < 16 * 128; i += 256) {
            int r = i >> 7;
            int c = (i & 127) * 8;                 // bf16 element offset
            const __bf16* src = hbuf + (size_t)s_t[r] * INTER + kp + c;
            __bf16*       dst = &sh[r * LDSTRIDE + c];
            __builtin_amdgcn_global_load_async_to_lds_b128(
                (v4i*)(void*)const_cast<__bf16*>(src),
                (v4i*)(void*)dst,
                0, 0);
        }
        wait_asynccnt0();
#else
        for (int i = threadIdx.x; i < 16 * 128; i += 256) {
            int r = i >> 7;
            int c = (i & 127) * 8;
            *(uint4*)&sh[r * LDSTRIDE + c] =
                *(const uint4*)(hbuf + (size_t)s_t[r] * INTER + kp + c);
        }
#endif
        __syncthreads();

        for (int k0 = 0; k0 < 1024; k0 += 32) {
            __builtin_prefetch(wr + kp + k0 + 32, 0, 1);
            v16bf a = load_a_frag(sh, lane, k0);
            v16bf b = load_b_frag_f32(wr + kp + k0);
            acc = __builtin_amdgcn_wmma_f32_16x16x32_bf16(false, a, false, b,
                                                          (short)0, acc, false, false);
        }
    }

    if (tile0 + 16 <= count) {
#pragma unroll
        for (int i = 0; i < 8; ++i)
            out[(size_t)trow[i] * DIM + d] = acc[i];
    } else {
#pragma unroll
        for (int i = 0; i < 8; ++i)
            if (tile0 + i + khalf * 8 < count)
                out[(size_t)trow[i] * DIM + d] = acc[i];
    }
}

// ---- launcher ---------------------------------------------------------------
extern "C" void kernel_launch(void* const* d_in, const int* in_sizes, int n_in,
                              void* d_out, int out_size, void* d_ws, size_t ws_size,
                              hipStream_t stream) {
    (void)in_sizes; (void)n_in; (void)out_size; (void)ws_size;

    const float* x    = (const float*)d_in[0];   // [M, D]
    const int*   eidx = (const int*)  d_in[1];   // [M, TOPK]
    const float* w13  = (const float*)d_in[2];   // [E, 2I, D]
    const float* w2   = (const float*)d_in[3];   // [E, D, I]
    float*       out  = (float*)d_out;           // [T, D]

    // workspace layout: counts(256B) | bucket(E*T ints = 64KB) | hbuf(T*I bf16 = 8MB)
    char*   ws     = (char*)d_ws;
    int*    counts = (int*)ws;
    int*    bucket = (int*)(ws + 256);
    __bf16* hbuf   = (__bf16*)(ws + 256 + (size_t)NEXP * TSLOTS * sizeof(int));

    moe_zero_counts<<<1, 32, 0, stream>>>(counts);
    moe_route<<<(TSLOTS + 255) / 256, 256, 0, stream>>>(eidx, counts, bucket);
    moe_gemm1<<<dim3(NEXP * 128, INTER / 128), 256, 0, stream>>>(x, bucket, counts, w13, hbuf);
    moe_gemm2<<<dim3(NEXP * 128, DIM / 128), 256, 0, stream>>>(hbuf, bucket, counts, w2, out);
}
